// ChannelAttentionModule_35459249995825
// MI455X (gfx1250) — matrix operand order repaired and checked by reading, not compiled, not run on hardware
//
#include <hip/hip_runtime.h>

// ---- CDNA5 WMMA types ----
typedef __attribute__((ext_vector_type(16))) __bf16 v16bf;
typedef __attribute__((ext_vector_type(8)))  float  v8f;

union Frag16 { uint4 q[2]; v16bf v; };

// f32 pair -> packed bf16 (RNE). Native v_cvt_pk_bf16_f32 when available.
__device__ __forceinline__ unsigned pack2bf16(float a, float b) {
#if __has_builtin(__builtin_amdgcn_cvt_pk_bf16_f32)
    auto r = __builtin_amdgcn_cvt_pk_bf16_f32(a, b);
    unsigned u; __builtin_memcpy(&u, &r, sizeof(u));
    return u;
#else
    unsigned ua = __builtin_bit_cast(unsigned, a);
    unsigned ub = __builtin_bit_cast(unsigned, b);
    ua = (ua + 0x7FFFu + ((ua >> 16) & 1u)) >> 16;
    ub = (ub + 0x7FFFu + ((ub >> 16) & 1u)) >> 16;
    return (ub << 16) | (ua & 0xFFFFu);
#endif
}

// interleave two words of packed bf16 pairs: a={a0,a1}, b={b0,b1} ->
// lo={a0,b0}, hi={a1,b1}  (k-pair packing for B fragments)
__device__ __forceinline__ void mix_pair(unsigned aw, unsigned bw,
                                         unsigned& lo, unsigned& hi) {
#if __has_builtin(__builtin_amdgcn_perm)
    lo = __builtin_amdgcn_perm(bw, aw, 0x05040100u);
    hi = __builtin_amdgcn_perm(bw, aw, 0x07060302u);
#else
    lo = (aw & 0xFFFFu) | (bw << 16);
    hi = (aw >> 16) | (bw & 0xFFFF0000u);
#endif
}

// ---- CDNA5 async LDS DMA (ASYNCcnt-tracked, no VGPR round-trip) ----
// GV mode: per-lane 16B  global -> LDS.  LDS offset = low 32 bits of the
// flat aperture address (see ISA 10.2).
__device__ __forceinline__ void async_ld_b128(void* lds_ptr, const void* gaddr) {
    unsigned lds_off = (unsigned)(size_t)lds_ptr;
    asm volatile("global_load_async_to_lds_b128 %0, %1, off"
                 :: "v"(lds_off), "v"(gaddr) : "memory");
}
__device__ __forceinline__ void wait_asynccnt0() {
#if __has_builtin(__builtin_amdgcn_s_wait_asynccnt)
    __builtin_amdgcn_s_wait_asynccnt(0);
#else
    asm volatile("s_wait_asynccnt 0" ::: "memory");
#endif
}

// LDS layout (bytes):
//   [0, 40960)      : union { stage x_b[512][40 bf16] (80B stride) |
//                             S[16][512] f32 | Bst[512][40 bf16] }
//   [40960, 57600)  : P bf16 [16][520] (1040B stride, 16B aligned)
//   [57600, 59648)  : reduction scratch, 2 x float[256]
#define LDS_P_OFF   40960
#define LDS_RED_OFF 57600
#define LDS_TOTAL   59648

// ===========================================================================
// One-time f32 -> bf16 conversion of x into workspace (bandwidth-bound pass)
// ===========================================================================
__global__ __launch_bounds__(256)
void cvt_to_bf16(const float* __restrict__ x, unsigned short* __restrict__ y, int n4) {
    int i = blockIdx.x * 256 + threadIdx.x;
    int stride = gridDim.x * 256;
    for (; i < n4; i += stride) {
        float4 v = ((const float4*)x)[i];
        uint2 p;
        p.x = pack2bf16(v.x, v.y);
        p.y = pack2bf16(v.z, v.w);
        ((uint2*)y)[i] = p;
    }
}

// ===========================================================================
// Fast path: fused kernel reading pre-converted bf16 (async-DMA staging)
// ===========================================================================
__global__ __launch_bounds__(256, 1)
void chanattn_fused_bf16(const unsigned short* __restrict__ xh,
                         const float* __restrict__ x,
                         const float* __restrict__ beta,
                         float* __restrict__ out) {
    __shared__ __align__(16) unsigned char lds[LDS_TOTAL];
    unsigned char* ldsU = lds;
    unsigned char* ldsP = lds + LDS_P_OFF;
    float* red  = (float*)(lds + LDS_RED_OFF);
    float* red2 = red + 256;

    const int t    = threadIdx.x;
    const int lane = t & 31;
    const int half = lane >> 4;
    const int l15  = lane & 15;
    const int wid  = t >> 5;
    const int wn0  = wid * 64;

    const int b  = blockIdx.x >> 5;
    const int r0 = (blockIdx.x & 31) * 16;
    const unsigned short* xbh = xh + (size_t)b * 512 * 4096;
    const float*          xb  = x  + (size_t)b * 512 * 4096;
    float*                ob  = out + (size_t)b * 512 * 4096;
    const float bet = beta[0];

    // ---------- Phase 1: S = x_b[r0:r0+16, :] @ x_b^T  (16 x 512, bf16 WMMA)
    v8f acc[4] = {};
    for (int kk = 0; kk < 4096; kk += 32) {
        __syncthreads();
        // stage 512 rows x 32 k bf16 via async LDS DMA (8 x B128 per thread)
        #pragma unroll
        for (int i = 0; i < 8; ++i) {
            int idx = t + i * 256;          // 0..2047 ; 4 uint4 per row
            int row = idx >> 2;
            int c   = idx & 3;
            async_ld_b128(ldsU + row * 80 + c * 16,
                          xbh + (size_t)row * 4096 + kk + c * 8);
        }
        // prefetch next chunk (2 rows per thread; rows are distinct lines)
        if (kk + 32 < 4096) {
            __builtin_prefetch(xbh + (size_t)(2 * t) * 4096 + kk + 32, 0, 3);
            __builtin_prefetch(xbh + (size_t)(2 * t + 1) * 4096 + kk + 32, 0, 3);
        }
        wait_asynccnt0();
        __syncthreads();
        // load A + all four B fragments first, then a back-to-back WMMA burst
        Frag16 fa, fb0, fb1, fb2, fb3;
        {
            const unsigned char* ab = ldsU + (r0 + l15) * 80 + half * 32;
            fa.q[0] = *(const uint4*)(ab);
            fa.q[1] = *(const uint4*)(ab + 16);
            const unsigned char* bb = ldsU + (wn0 + l15) * 80 + half * 32;
            fb0.q[0] = *(const uint4*)(bb);        fb0.q[1] = *(const uint4*)(bb + 16);
            fb1.q[0] = *(const uint4*)(bb + 1280); fb1.q[1] = *(const uint4*)(bb + 1296);
            fb2.q[0] = *(const uint4*)(bb + 2560); fb2.q[1] = *(const uint4*)(bb + 2576);
            fb3.q[0] = *(const uint4*)(bb + 3840); fb3.q[1] = *(const uint4*)(bb + 3856);
        }
        acc[0] = __builtin_amdgcn_wmma_f32_16x16x32_bf16(false, fa.v, false, fb0.v, (short)0, acc[0], false, false);
        acc[1] = __builtin_amdgcn_wmma_f32_16x16x32_bf16(false, fa.v, false, fb1.v, (short)0, acc[1], false, false);
        acc[2] = __builtin_amdgcn_wmma_f32_16x16x32_bf16(false, fa.v, false, fb2.v, (short)0, acc[2], false, false);
        acc[3] = __builtin_amdgcn_wmma_f32_16x16x32_bf16(false, fa.v, false, fb3.v, (short)0, acc[3], false, false);
    }

    // ---------- Phase 2: softmax(rowmin - S) -> P (bf16 in LDS)
    __syncthreads();
    float* S = (float*)ldsU;
    #pragma unroll
    for (int tn = 0; tn < 4; ++tn)
        #pragma unroll
        for (int v = 0; v < 8; ++v)
            S[(v + half * 8) * 512 + wn0 + tn * 16 + l15] = acc[tn][v];
    __syncthreads();
    {
        int row = t >> 4;
        int seg = (t & 15) * 32;
        const float* srow = S + row * 512 + seg;
        float mn = srow[0];
        #pragma unroll
        for (int i = 1; i < 32; ++i) mn = fminf(mn, srow[i]);
        red[t] = mn;
        __syncthreads();
        float rmn = red[row * 16];
        #pragma unroll
        for (int i = 1; i < 16; ++i) rmn = fminf(rmn, red[row * 16 + i]);
        float e[32]; float s = 0.f;
        #pragma unroll
        for (int i = 0; i < 32; ++i) { e[i] = __expf(rmn - srow[i]); s += e[i]; }
        red2[t] = s;
        __syncthreads();
        float rs = 0.f;
        #pragma unroll
        for (int i = 0; i < 16; ++i) rs += red2[row * 16 + i];
        float inv = 1.f / rs;
        unsigned* prow = (unsigned*)(ldsP + row * 1040) + (seg >> 1);
        #pragma unroll
        for (int i = 0; i < 16; ++i)
            prow[i] = pack2bf16(e[2 * i] * inv, e[2 * i + 1] * inv);
    }

    // ---------- Phase 3: E = P @ x_b ; out = beta*E + x
    for (int n0 = 0; n0 < 4096; n0 += 512) {
        v8f eacc[4] = {};
        for (int k0 = 0; k0 < 512; k0 += 32) {
            __syncthreads();
            // stage Bst[c][k] (transposed, k-pairs packed) from bf16 source
            #pragma unroll
            for (int i = 0; i < 4; ++i) {
                int idx = t + i * 256;       // 0..1023 ; 16 pr x 64 colgroup8
                int pr  = idx >> 6;
                int c8  = idx & 63;
                const unsigned short* p0 = xbh + (size_t)(k0 + 2 * pr) * 4096 + n0 + c8 * 8;
                uint4 a = *(const uint4*)p0;
                uint4 bb = *(const uint4*)(p0 + 4096);
                unsigned char* d = ldsU + (c8 * 8) * 80 + pr * 4;
                unsigned lo, hi;
                mix_pair(a.x, bb.x, lo, hi);
                *(unsigned*)(d +   0) = lo; *(unsigned*)(d +  80) = hi;
                mix_pair(a.y, bb.y, lo, hi);
                *(unsigned*)(d + 160) = lo; *(unsigned*)(d + 240) = hi;
                mix_pair(a.z, bb.z, lo, hi);
                *(unsigned*)(d + 320) = lo; *(unsigned*)(d + 400) = hi;
                mix_pair(a.w, bb.w, lo, hi);
                *(unsigned*)(d + 480) = lo; *(unsigned*)(d + 560) = hi;
            }
            __syncthreads();
            Frag16 fa, fb0, fb1, fb2, fb3;
            {
                const unsigned char* ab = ldsP + l15 * 1040 + k0 * 2 + half * 32;
                fa.q[0] = *(const uint4*)(ab);
                fa.q[1] = *(const uint4*)(ab + 16);
                const unsigned char* bb = ldsU + (wn0 + l15) * 80 + half * 32;
                fb0.q[0] = *(const uint4*)(bb);        fb0.q[1] = *(const uint4*)(bb + 16);
                fb1.q[0] = *(const uint4*)(bb + 1280); fb1.q[1] = *(const uint4*)(bb + 1296);
                fb2.q[0] = *(const uint4*)(bb + 2560); fb2.q[1] = *(const uint4*)(bb + 2576);
                fb3.q[0] = *(const uint4*)(bb + 3840); fb3.q[1] = *(const uint4*)(bb + 3856);
            }
            eacc[0] = __builtin_amdgcn_wmma_f32_16x16x32_bf16(false, fa.v, false, fb0.v, (short)0, eacc[0], false, false);
            eacc[1] = __builtin_amdgcn_wmma_f32_16x16x32_bf16(false, fa.v, false, fb1.v, (short)0, eacc[1], false, false);
            eacc[2] = __builtin_amdgcn_wmma_f32_16x16x32_bf16(false, fa.v, false, fb2.v, (short)0, eacc[2], false, false);
            eacc[3] = __builtin_amdgcn_wmma_f32_16x16x32_bf16(false, fa.v, false, fb3.v, (short)0, eacc[3], false, false);
        }
        #pragma unroll
        for (int tn = 0; tn < 4; ++tn)
            #pragma unroll
            for (int v = 0; v < 8; ++v) {
                int row = r0 + v + half * 8;
                int col = n0 + wn0 + tn * 16 + l15;
                size_t off = (size_t)row * 4096 + col;
                ob[off] = bet * eacc[tn][v] + xb[off];
            }
    }
}

// ===========================================================================
// Fallback: fully fused fp32-input kernel (used when workspace is too small)
// ===========================================================================
__global__ __launch_bounds__(256, 1)
void chanattn_fused(const float* __restrict__ x, const float* __restrict__ beta,
                    float* __restrict__ out) {
    __shared__ __align__(16) unsigned char lds[LDS_TOTAL];
    unsigned char* ldsU = lds;
    unsigned char* ldsP = lds + LDS_P_OFF;
    float* red  = (float*)(lds + LDS_RED_OFF);
    float* red2 = red + 256;

    const int t    = threadIdx.x;
    const int lane = t & 31;
    const int half = lane >> 4;
    const int l15  = lane & 15;
    const int wid  = t >> 5;
    const int wn0  = wid * 64;

    const int b  = blockIdx.x >> 5;
    const int r0 = (blockIdx.x & 31) * 16;
    const float* xb = x   + (size_t)b * 512 * 4096;
    float*       ob = out + (size_t)b * 512 * 4096;
    const float bet = beta[0];

    v8f acc[4] = {};
    for (int kk = 0; kk < 4096; kk += 32) {
        __syncthreads();
        #pragma unroll
        for (int i = 0; i < 16; ++i) {
            int idx = t + i * 256;
            int row = idx >> 3;
            int c4  = idx & 7;
            float4 v = *(const float4*)(xb + (size_t)row * 4096 + kk + c4 * 4);
            uint2 pk;
            pk.x = pack2bf16(v.x, v.y);
            pk.y = pack2bf16(v.z, v.w);
            *(uint2*)(ldsU + row * 80 + c4 * 8) = pk;
        }
        __syncthreads();
        Frag16 fa;
        {
            const unsigned char* ab = ldsU + (r0 + l15) * 80 + half * 32;
            fa.q[0] = *(const uint4*)(ab);
            fa.q[1] = *(const uint4*)(ab + 16);
        }
        #pragma unroll
        for (int tn = 0; tn < 4; ++tn) {
            Frag16 fb;
            const unsigned char* bb = ldsU + (wn0 + tn * 16 + l15) * 80 + half * 32;
            fb.q[0] = *(const uint4*)(bb);
            fb.q[1] = *(const uint4*)(bb + 16);
            acc[tn] = __builtin_amdgcn_wmma_f32_16x16x32_bf16(
                false, fa.v, false, fb.v, (short)0, acc[tn], false, false);
        }
    }

    __syncthreads();
    float* S = (float*)ldsU;
    #pragma unroll
    for (int tn = 0; tn < 4; ++tn)
        #pragma unroll
        for (int v = 0; v < 8; ++v)
            S[(v + half * 8) * 512 + wn0 + tn * 16 + l15] = acc[tn][v];
    __syncthreads();
    {
        int row = t >> 4;
        int seg = (t & 15) * 32;
        const float* srow = S + row * 512 + seg;
        float mn = srow[0];
        #pragma unroll
        for (int i = 1; i < 32; ++i) mn = fminf(mn, srow[i]);
        red[t] = mn;
        __syncthreads();
        float rmn = red[row * 16];
        #pragma unroll
        for (int i = 1; i < 16; ++i) rmn = fminf(rmn, red[row * 16 + i]);
        float e[32]; float s = 0.f;
        #pragma unroll
        for (int i = 0; i < 32; ++i) { e[i] = __expf(rmn - srow[i]); s += e[i]; }
        red2[t] = s;
        __syncthreads();
        float rs = 0.f;
        #pragma unroll
        for (int i = 0; i < 16; ++i) rs += red2[row * 16 + i];
        float inv = 1.f / rs;
        unsigned* prow = (unsigned*)(ldsP + row * 1040) + (seg >> 1);
        #pragma unroll
        for (int i = 0; i < 16; ++i)
            prow[i] = pack2bf16(e[2 * i] * inv, e[2 * i + 1] * inv);
    }

    for (int n0 = 0; n0 < 4096; n0 += 512) {
        v8f eacc[4] = {};
        for (int k0 = 0; k0 < 512; k0 += 32) {
            __syncthreads();
            #pragma unroll
            for (int i = 0; i < 8; ++i) {
                int idx = t + i * 256;
                int pr  = idx >> 7;
                int c4  = idx & 127;
                const float* p0 = xb + (size_t)(k0 + 2 * pr) * 4096 + n0 + c4 * 4;
                float4 va = *(const float4*)p0;
                float4 vb = *(const float4*)(p0 + 4096);
                unsigned char* d = ldsU + (c4 * 4) * 80 + pr * 4;
                *(unsigned*)(d)       = pack2bf16(va.x, vb.x);
                *(unsigned*)(d +  80) = pack2bf16(va.y, vb.y);
                *(unsigned*)(d + 160) = pack2bf16(va.z, vb.z);
                *(unsigned*)(d + 240) = pack2bf16(va.w, vb.w);
            }
            __syncthreads();
            Frag16 fa;
            {
                const unsigned char* ab = ldsP + l15 * 1040 + k0 * 2 + half * 32;
                fa.q[0] = *(const uint4*)(ab);
                fa.q[1] = *(const uint4*)(ab + 16);
            }
            #pragma unroll
            for (int tn = 0; tn < 4; ++tn) {
                Frag16 fb;
                const unsigned char* bb = ldsU + (wn0 + tn * 16 + l15) * 80 + half * 32;
                fb.q[0] = *(const uint4*)(bb);
                fb.q[1] = *(const uint4*)(bb + 16);
                eacc[tn] = __builtin_amdgcn_wmma_f32_16x16x32_bf16(
                    false, fa.v, false, fb.v, (short)0, eacc[tn], false, false);
            }
        }
        #pragma unroll
        for (int tn = 0; tn < 4; ++tn)
            #pragma unroll
            for (int v = 0; v < 8; ++v) {
                int row = r0 + v + half * 8;
                int col = n0 + wn0 + tn * 16 + l15;
                size_t off = (size_t)row * 4096 + col;
                ob[off] = bet * eacc[tn][v] + xb[off];
            }
    }
}

extern "C" void kernel_launch(void* const* d_in, const int* in_sizes, int n_in,
                              void* d_out, int out_size, void* d_ws, size_t ws_size,
                              hipStream_t stream) {
    const float* x    = (const float*)d_in[0];
    const float* beta = (const float*)d_in[1];
    float* out = (float*)d_out;

    const size_t n_elems = (size_t)16 * 512 * 4096;
    if (ws_size >= n_elems * sizeof(unsigned short)) {
        // pass 1: one-time bf16 conversion into workspace
        unsigned short* xh = (unsigned short*)d_ws;
        cvt_to_bf16<<<dim3(2048), dim3(256), 0, stream>>>(x, xh, (int)(n_elems / 4));
        // pass 2: fused attention (async-DMA bf16 staging)
        chanattn_fused_bf16<<<dim3(512), dim3(256), 0, stream>>>(xh, x, beta, out);
    } else {
        chanattn_fused<<<dim3(512), dim3(256), 0, stream>>>(x, beta, out);
    }
}
